// UniMixerLiteLayer_13443247637018
// MI455X (gfx1250) — matrix operand order repaired and checked by reading, not compiled, not run on hardware
//
#include <hip/hip_runtime.h>
#include <hip/hip_bf16.h>
#include <math.h>

// ---------------------------------------------------------------------------
// UniMixerLite layer for MI455X (gfx1250, wave32, WMMA)
// B=64, T=64, D=256, N=K=128, BASIS=8, F_FF=1024
// ---------------------------------------------------------------------------

typedef __attribute__((ext_vector_type(8)))  float  v8f;
typedef __attribute__((ext_vector_type(2)))  float  v2f;
typedef __attribute__((ext_vector_type(16))) __bf16 v16bf;

#define SINK_ITERS 20
#define INV_TEMP   5.0f     // 1/0.2
#define LN_EPS     1e-5f

// ---------------- small helpers ----------------

static __device__ __forceinline__ float waveSum(float v) {
#pragma unroll
  for (int off = 16; off > 0; off >>= 1) v += __shfl_xor(v, off, 32);
  return v;
}

static __device__ __forceinline__ __bf16 f2bf(float f) { return (__bf16)f; }

static __device__ __forceinline__ v8f zero8() {
  v8f z;
#pragma unroll
  for (int i = 0; i < 8; ++i) z[i] = 0.0f;
  return z;
}

// block-wide mean / inv-std over 256 values (one per thread), 8 waves
static __device__ __forceinline__ void blockMeanInv(float v, int tid, float* red,
                                                    float& mean, float& inv) {
  float s  = waveSum(v);
  float s2 = waveSum(v * v);
  int w = tid >> 5;
  if ((tid & 31) == 0) { red[w] = s; red[8 + w] = s2; }
  __syncthreads();
  if (tid == 0) {
    float a = 0.f, b = 0.f;
#pragma unroll
    for (int i = 0; i < 8; ++i) { a += red[i]; b += red[8 + i]; }
    float mu  = a * (1.0f / 256.0f);
    float var = b * (1.0f / 256.0f) - mu * mu;
    red[16] = mu;
    red[17] = rsqrtf(var + LN_EPS);
  }
  __syncthreads();
  mean = red[16];
  inv  = red[17];
  __syncthreads();   // safe for repeated use of `red`
}

// LDS swizzles (bank-conflict free, keep buffers at exactly 64KB)
static __device__ __forceinline__ int swz128(int r, int c) {   // 128x128 f32
  return r * 128 + (c ^ ((r << 1) & 63));
}
static __device__ __forceinline__ int swz256(int m, int k) {   // 64x256 f32
  return m * 256 + (k ^ ((m << 2) & 63));
}

// ---------------- WMMA fragment loaders ----------------

// A fragment for V_WMMA_F32_16X16X32_BF16. A is 16(M) x 32(K), row-major, ld in elems.
static __device__ __forceinline__ v16bf loadA_bf16(const __bf16* A, int ld, int m,
                                                   int k0, int lane) {
  int half = lane >> 4;
  const __bf16* p = A + m * ld + k0 + half * 8;
  v16bf a;
#pragma unroll
  for (int i = 0; i < 8; ++i) a[i] = p[i];
#pragma unroll
  for (int i = 0; i < 8; ++i) a[8 + i] = p[16 + i];
  return a;
}

// B fragment from *transposed* storage Bt[n][k] (so B[k][n]=Bt[n][k]); same k order as A (two 8-elem runs).
static __device__ __forceinline__ v16bf loadBt_bf16(const __bf16* Bt, int ld, int n,
                                                    int k0, int lane) {
  int half = lane >> 4;
  const __bf16* p = Bt + n * ld + k0 + half * 8;
  v16bf b;
#pragma unroll
  for (int i = 0; i < 8; ++i) b[i] = p[i];
#pragma unroll
  for (int i = 0; i < 8; ++i) b[8 + i] = p[16 + i];
  return b;
}

static __device__ __forceinline__ v8f wmma_bf16(v16bf a, v16bf b, v8f c) {
  return __builtin_amdgcn_wmma_f32_16x16x32_bf16(false, a, false, b, (short)0, c,
                                                 false, false);
}
static __device__ __forceinline__ v8f wmma_f32(v2f a, v2f b, v8f c) {
  return __builtin_amdgcn_wmma_f32_16x16x4_f32(false, a, false, b, (short)0, c,
                                               false, false);
}

// ---------------------------------------------------------------------------
// Kernel 1: Sinkhorn on A_logits [128,128] -> At (bf16, transposed: At[n'][n])
// ---------------------------------------------------------------------------
__global__ void k_sinkhorn_A(const float* __restrict__ logits,
                             __bf16* __restrict__ At) {
  __shared__ float la[128 * 128];   // 64KB, swizzled
  int tid = threadIdx.x;            // 128 threads
  for (int idx = tid; idx < 128 * 128; idx += 128) {
    int r = idx >> 7, c = idx & 127;
    la[swz128(r, c)] = logits[idx] * INV_TEMP;
  }
  __syncthreads();
  for (int it = 0; it < SINK_ITERS; ++it) {
    {  // rows (axis = -1)
      int r = tid;
      float m = -3.402823466e38f;
      for (int c = 0; c < 128; ++c) m = fmaxf(m, la[swz128(r, c)]);
      float s = 0.f;
      for (int c = 0; c < 128; ++c) s += __expf(la[swz128(r, c)] - m);
      float l = m + __logf(s);
      for (int c = 0; c < 128; ++c) la[swz128(r, c)] -= l;
    }
    __syncthreads();
    {  // cols (axis = -2)
      int c = tid;
      float m = -3.402823466e38f;
      for (int r = 0; r < 128; ++r) m = fmaxf(m, la[swz128(r, c)]);
      float s = 0.f;
      for (int r = 0; r < 128; ++r) s += __expf(la[swz128(r, c)] - m);
      float l = m + __logf(s);
      for (int r = 0; r < 128; ++r) la[swz128(r, c)] -= l;
    }
    __syncthreads();
  }
  // At[c][r] = A[r][c]
  for (int q = tid; q < 128 * 128; q += 128) {
    int c = q >> 7, r = q & 127;
    At[q] = f2bf(__expf(la[swz128(r, c)]));
  }
}

// ---------------------------------------------------------------------------
// Kernel 2: W_logits[n] = W_1[n,:] . W_V, Sinkhorn, emit Wt[n][o][i] (bf16)
// ---------------------------------------------------------------------------
__global__ void k_sinkhorn_W(const float* __restrict__ W1,
                             const float* __restrict__ WV,
                             __bf16* __restrict__ Wt) {
  __shared__ float lw[128 * 128];   // 64KB, swizzled
  int n = blockIdx.x, tid = threadIdx.x;   // 256 threads
  float w1[8];
#pragma unroll
  for (int k = 0; k < 8; ++k) w1[k] = W1[n * 8 + k];
  for (int idx = tid; idx < 16384; idx += 256) {
    float acc = 0.f;
#pragma unroll
    for (int k = 0; k < 8; ++k) acc += w1[k] * WV[k * 16384 + idx];
    lw[swz128(idx >> 7, idx & 127)] = acc * INV_TEMP;
  }
  __syncthreads();
  for (int it = 0; it < SINK_ITERS; ++it) {
    if (tid < 128) {
      int r = tid;
      float m = -3.402823466e38f;
      for (int c = 0; c < 128; ++c) m = fmaxf(m, lw[swz128(r, c)]);
      float s = 0.f;
      for (int c = 0; c < 128; ++c) s += __expf(lw[swz128(r, c)] - m);
      float l = m + __logf(s);
      for (int c = 0; c < 128; ++c) lw[swz128(r, c)] -= l;
    }
    __syncthreads();
    if (tid < 128) {
      int c = tid;
      float m = -3.402823466e38f;
      for (int r = 0; r < 128; ++r) m = fmaxf(m, lw[swz128(r, c)]);
      float s = 0.f;
      for (int r = 0; r < 128; ++r) s += __expf(lw[swz128(r, c)] - m);
      float l = m + __logf(s);
      for (int r = 0; r < 128; ++r) lw[swz128(r, c)] -= l;
    }
    __syncthreads();
  }
  // Wt[n][o][i] = W[n][i][o]   (transposed for contiguous WMMA-B fragments)
  __bf16* out = Wt + n * 16384;
  for (int q = tid; q < 16384; q += 256) {
    int o = q >> 7, i = q & 127;
    out[q] = f2bf(__expf(lw[swz128(i, o)]));
  }
}

// ---------------------------------------------------------------------------
// Kernel 3: mixing_input = src_x + LN(src_y); token-mix permute; cast -> xr bf16
//           xr[b][n][k] with flat(n,k) = t2*256 + t1*4 + c, d = t2*4+c
// ---------------------------------------------------------------------------
__global__ void k_premix(const float* __restrict__ sx, const float* __restrict__ sy,
                         const float* __restrict__ g, const float* __restrict__ b,
                         __bf16* __restrict__ xr) {
  __shared__ float red[32];
  int bt = blockIdx.x;               // b*64 + t
  int t  = bt & 63;
  int d  = threadIdx.x;              // 256 threads
  int base = bt * 256;
  float y = sy[base + d];
  float mean, inv;
  blockMeanInv(y, d, red, mean, inv);
  float m = sx[base + d] + (y - mean) * inv * g[d] + b[d];
  int j = d >> 2, c = d & 3;
  xr[(bt >> 6) * 16384 + j * 256 + t * 4 + c] = f2bf(m);
}

// ---------------------------------------------------------------------------
// Kernel 4: x_local[b][n][o] = sum_i xr[b][n][i] * W[n][i][o]  (bf16 WMMA)
//           one block per n; stores transposed: xlT[b][o][n]
// ---------------------------------------------------------------------------
__global__ void __launch_bounds__(256) k_xlocal(const __bf16* __restrict__ xr,
                                                const __bf16* __restrict__ Wt,
                                                float* __restrict__ xlT) {
  int n = blockIdx.x;
  int tid = threadIdx.x, wave = tid >> 5, lane = tid & 31;
  int cl = lane & 15, half = lane >> 4;
  const __bf16* Wn    = Wt + n * 16384;   // Wt[n][o][i]
  const __bf16* Abase = xr + n * 128;     // A[m][i] = xr[m*16384 + n*128 + i]
  v8f acc[4];
#pragma unroll
  for (int i = 0; i < 4; ++i) acc[i] = zero8();
#pragma unroll
  for (int ks = 0; ks < 4; ++ks) {
    int k0 = ks * 32;
    v16bf bfrag = loadBt_bf16(Wn, 128, wave * 16 + cl, k0, lane);
#pragma unroll
    for (int mt = 0; mt < 4; ++mt) {
      v16bf afrag = loadA_bf16(Abase, 16384, mt * 16 + cl, k0, lane);
      acc[mt] = wmma_bf16(afrag, bfrag, acc[mt]);
    }
  }
  int o = wave * 16 + cl;
#pragma unroll
  for (int mt = 0; mt < 4; ++mt)
#pragma unroll
    for (int r = 0; r < 8; ++r) {
      int brow = mt * 16 + half * 8 + r;
      xlT[brow * 16384 + o * 128 + n] = acc[mt][r];
    }
}

// ---------------------------------------------------------------------------
// Kernel 5: LayerNorm over n (128) per (b,k) row; cast -> xln bf16 [b][k][n]
// ---------------------------------------------------------------------------
__global__ void k_ln_n(const float* __restrict__ xlT, const float* __restrict__ g,
                       const float* __restrict__ b, __bf16* __restrict__ xln) {
  int tid = threadIdx.x, wave = tid >> 5, lane = tid & 31;
  int row = blockIdx.x * 8 + wave;   // over B*K = 8192 rows
  const float* p = xlT + row * 128;
  float v[4], s = 0.f, s2 = 0.f;
#pragma unroll
  for (int j = 0; j < 4; ++j) {
    v[j] = p[lane + 32 * j];
    s += v[j]; s2 += v[j] * v[j];
  }
  s = waveSum(s); s2 = waveSum(s2);
  float mu  = s * (1.f / 128.f);
  float inv = rsqrtf(s2 * (1.f / 128.f) - mu * mu + LN_EPS);
  __bf16* q = xln + row * 128;
#pragma unroll
  for (int j = 0; j < 4; ++j) {
    int nidx = lane + 32 * j;
    q[nidx] = f2bf((v[j] - mu) * inv * g[nidx] + b[nidx]);
  }
}

// ---------------------------------------------------------------------------
// Kernel 6: x_global = xln[b] @ A  (bf16 WMMA), stored directly as mixing_h
// ---------------------------------------------------------------------------
__global__ void __launch_bounds__(256) k_xglobal(const __bf16* __restrict__ xln,
                                                 const __bf16* __restrict__ At,
                                                 float* __restrict__ mh) {
  int b = blockIdx.x;
  int tid = threadIdx.x, wave = tid >> 5, lane = tid & 31;
  int cl = lane & 15, half = lane >> 4;
  const __bf16* Ab = xln + b * 16384;   // [k][n], ld=128
  v8f acc[8];
#pragma unroll
  for (int i = 0; i < 8; ++i) acc[i] = zero8();
#pragma unroll
  for (int ks = 0; ks < 4; ++ks) {
    int k0 = ks * 32;
    v16bf afrag = loadA_bf16(Ab, 128, wave * 16 + cl, k0, lane);
#pragma unroll
    for (int nt = 0; nt < 8; ++nt) {
      v16bf bfrag = loadBt_bf16(At, 128, nt * 16 + cl, k0, lane);
      acc[nt] = wmma_bf16(afrag, bfrag, acc[nt]);
    }
  }
  float* out = mh + b * 16384;
#pragma unroll
  for (int nt = 0; nt < 8; ++nt) {
    int np = nt * 16 + cl;
#pragma unroll
    for (int r = 0; r < 8; ++r) {
      int krow = wave * 16 + half * 8 + r;
      out[np * 128 + krow] = acc[nt][r];   // mixing_h flat = n'*128 + k
    }
  }
}

// ---------------------------------------------------------------------------
// Kernel 7: mid_x = LN(src_x+mh); mid_y = src_y+mh; pffn = mid_x + LN(mid_y)
// ---------------------------------------------------------------------------
__global__ void k_residual(const float* __restrict__ sx, const float* __restrict__ sy,
                           const float* __restrict__ mh,
                           const float* __restrict__ g1, const float* __restrict__ b1,
                           const float* __restrict__ g2, const float* __restrict__ b2,
                           float* __restrict__ midx, float* __restrict__ midy,
                           float* __restrict__ pffn) {
  __shared__ float red[32];
  int bt = blockIdx.x, d = threadIdx.x;
  int base = bt * 256;
  float h = mh[base + d];
  float s = sx[base + d] + h;
  float mean, inv;
  blockMeanInv(s, d, red, mean, inv);
  float mx = (s - mean) * inv * g1[d] + b1[d];
  float my = sy[base + d] + h;
  blockMeanInv(my, d, red, mean, inv);
  float pf = mx + (my - mean) * inv * g2[d] + b2[d];
  midx[base + d] = mx;
  midy[base + d] = my;
  pffn[base + d] = pf;
}

// ---------------------------------------------------------------------------
// Kernel 8: FFN part 1 — gate/up (f32 WMMA 16x16x4, weights streamed as f32),
//           hidden = silu(gate)*up.  grid = (ftile=8, l=64)
// ---------------------------------------------------------------------------
__global__ void __launch_bounds__(256) k_ffn1(const float* __restrict__ pffn,
                                              const float* __restrict__ Wg,
                                              const float* __restrict__ Wu,
                                              float* __restrict__ hid) {
  __shared__ float As[64 * 256];   // 64KB, swizzled
  int l  = blockIdx.y;
  int f0 = blockIdx.x * 128;
  int tid = threadIdx.x;
  for (int idx = tid; idx < 64 * 256; idx += 256) {
    int r = idx >> 8, c = idx & 255;
    As[swz256(r, c)] = pffn[(r * 64 + l) * 256 + c];
  }
  __syncthreads();
  int wave = tid >> 5, lane = tid & 31, cl = lane & 15, half = lane >> 4;
  int fc = f0 + wave * 16 + cl;
  v8f ag[4], au[4];
#pragma unroll
  for (int i = 0; i < 4; ++i) { ag[i] = zero8(); au[i] = zero8(); }
  for (int ks = 0; ks < 64; ++ks) {
    int kA = ks * 4 + half * 2;
    const float* pg = Wg + (l * 256 + kA) * 1024 + fc;
    const float* pu = Wu + (l * 256 + kA) * 1024 + fc;
    v2f bg; bg[0] = pg[0]; bg[1] = pg[1024];
    v2f bu; bu[0] = pu[0]; bu[1] = pu[1024];
#pragma unroll
    for (int mt = 0; mt < 4; ++mt) {
      int m = mt * 16 + cl;
      v2f a; a[0] = As[swz256(m, kA)]; a[1] = As[swz256(m, kA + 1)];
      ag[mt] = wmma_f32(a, bg, ag[mt]);
      au[mt] = wmma_f32(a, bu, au[mt]);
    }
  }
#pragma unroll
  for (int mt = 0; mt < 4; ++mt)
#pragma unroll
    for (int r = 0; r < 8; ++r) {
      int brow = mt * 16 + half * 8 + r;
      float gv = ag[mt][r], uv = au[mt][r];
      float hv = gv * uv / (1.f + __expf(-gv));   // silu(g)*u
      hid[(brow * 64 + l) * 1024 + fc] = hv;
    }
}

// ---------------------------------------------------------------------------
// Kernel 9: FFN part 2 — down-proj (f32 WMMA) + fused final LN / residuals.
//           grid = 64 (one block per token l)
// ---------------------------------------------------------------------------
__global__ void __launch_bounds__(256) k_ffn2(const float* __restrict__ hid,
                                              const float* __restrict__ Wd,
                                              const float* __restrict__ midx,
                                              const float* __restrict__ midy,
                                              const float* __restrict__ g,
                                              const float* __restrict__ beta,
                                              float* __restrict__ outx,
                                              float* __restrict__ outy) {
  __shared__ float Hs[64 * 256];   // 64KB, swizzled; reused for output staging
  int l = blockIdx.x;
  int tid = threadIdx.x, wave = tid >> 5, lane = tid & 31;
  int cl = lane & 15, half = lane >> 4;
  v8f acc[8];
#pragma unroll
  for (int i = 0; i < 8; ++i) acc[i] = zero8();
  for (int chunk = 0; chunk < 4; ++chunk) {
    int kb = chunk * 256;
    __syncthreads();
    for (int idx = tid; idx < 64 * 256; idx += 256) {
      int r = idx >> 8, c = idx & 255;
      Hs[swz256(r, c)] = hid[(r * 64 + l) * 1024 + kb + c];
    }
    __syncthreads();
    for (int ks = 0; ks < 64; ++ks) {
      int kA = ks * 4 + half * 2;
      int d0 = (wave * 2 + 0) * 16 + cl;
      int d1 = (wave * 2 + 1) * 16 + cl;
      const float* p0 = Wd + (l * 1024 + kb + kA) * 256 + d0;
      const float* p1 = Wd + (l * 1024 + kb + kA) * 256 + d1;
      v2f b0; b0[0] = p0[0]; b0[1] = p0[256];
      v2f b1; b1[0] = p1[0]; b1[1] = p1[256];
#pragma unroll
      for (int mt = 0; mt < 4; ++mt) {
        int m = mt * 16 + cl;
        v2f a; a[0] = Hs[swz256(m, kA)]; a[1] = Hs[swz256(m, kA + 1)];
        acc[mt * 2 + 0] = wmma_f32(a, b0, acc[mt * 2 + 0]);
        acc[mt * 2 + 1] = wmma_f32(a, b1, acc[mt * 2 + 1]);
      }
    }
  }
  __syncthreads();
  // stage ffn_out tile [64 x 256] into LDS
#pragma unroll
  for (int mt = 0; mt < 4; ++mt)
#pragma unroll
    for (int nti = 0; nti < 2; ++nti) {
      int d = (wave * 2 + nti) * 16 + cl;
#pragma unroll
      for (int r = 0; r < 8; ++r) {
        int brow = mt * 16 + half * 8 + r;
        Hs[swz256(brow, d)] = acc[mt * 2 + nti][r];
      }
    }
  __syncthreads();
  // fused epilogue: out_x = LN(mid_x + ffn_out), out_y = mid_y + ffn_out
  for (int row = wave * 8; row < wave * 8 + 8; ++row) {
    int base = (row * 64 + l) * 256;
    float sv[8], fv[8], s = 0.f, s2 = 0.f;
#pragma unroll
    for (int j = 0; j < 8; ++j) {
      int d = lane + 32 * j;
      float f = Hs[swz256(row, d)];
      float x = midx[base + d] + f;
      sv[j] = x; fv[j] = f;
      s += x; s2 += x * x;
    }
    s = waveSum(s); s2 = waveSum(s2);
    float mu  = s * (1.f / 256.f);
    float inv = rsqrtf(s2 * (1.f / 256.f) - mu * mu + LN_EPS);
#pragma unroll
    for (int j = 0; j < 8; ++j) {
      int d = lane + 32 * j;
      outx[base + d] = (sv[j] - mu) * inv * g[d] + beta[d];
      outy[base + d] = midy[base + d] + fv[j];
    }
  }
}

// ---------------------------------------------------------------------------
// Host launch
// ---------------------------------------------------------------------------
extern "C" void kernel_launch(void* const* d_in, const int* in_sizes, int n_in,
                              void* d_out, int out_size, void* d_ws, size_t ws_size,
                              hipStream_t stream) {
  (void)in_sizes; (void)n_in; (void)out_size; (void)ws_size;

  const float* src_x      = (const float*)d_in[0];
  const float* src_y      = (const float*)d_in[1];
  const float* A_logits   = (const float*)d_in[2];
  const float* W_1        = (const float*)d_in[3];
  const float* W_V        = (const float*)d_in[4];
  const float* norm2_g    = (const float*)d_in[5];
  const float* norm2_b    = (const float*)d_in[6];
  const float* W_gate     = (const float*)d_in[7];
  const float* W_up       = (const float*)d_in[8];
  const float* W_down     = (const float*)d_in[9];
  const float* ln_premix_g  = (const float*)d_in[10];
  const float* ln_premix_b  = (const float*)d_in[11];
  const float* ln_postmix_g = (const float*)d_in[12];
  const float* ln_postmix_b = (const float*)d_in[13];
  const float* ln_preffn_g  = (const float*)d_in[14];
  const float* ln_preffn_b  = (const float*)d_in[15];
  const float* ln_postmoe_g = (const float*)d_in[16];
  const float* ln_postmoe_b = (const float*)d_in[17];

  char* ws = (char*)d_ws;
  // workspace layout (bytes)
  __bf16* At   = (__bf16*)(ws + 0);                 //   32 KB  A^T bf16
  __bf16* Wt   = (__bf16*)(ws + 32768);             //    4 MB  W^T bf16 [n][o][i]
  __bf16* xr   = (__bf16*)(ws + 4227072);           //    2 MB  xr bf16 [b][n][k]
  float*  xlT  = (float*) (ws + 6324224);           //    4 MB  x_local^T [b][k][n]
  __bf16* xln  = (__bf16*)(ws + 10518528);          //    2 MB  LN(x_local^T) bf16
  float*  mh   = (float*) (ws + 12615680);          //    4 MB  mixing_h
  float*  midx = (float*) (ws + 16809984);          //    4 MB
  float*  midy = (float*) (ws + 21004288);          //    4 MB
  float*  pffn = (float*) (ws + 25198592);          //    4 MB
  float*  hid  = (float*) (ws + 29392896);          //   16 MB  hidden [b][l][f]

  float* outx = (float*)d_out;
  float* outy = outx + 64 * 64 * 256;

  k_sinkhorn_A<<<1, 128, 0, stream>>>(A_logits, At);
  k_sinkhorn_W<<<128, 256, 0, stream>>>(W_1, W_V, Wt);
  k_premix<<<4096, 256, 0, stream>>>(src_x, src_y, ln_premix_g, ln_premix_b, xr);
  k_xlocal<<<128, 256, 0, stream>>>(xr, Wt, xlT);
  k_ln_n<<<1024, 256, 0, stream>>>(xlT, norm2_g, norm2_b, xln);
  k_xglobal<<<64, 256, 0, stream>>>(xln, At, mh);
  k_residual<<<4096, 256, 0, stream>>>(src_x, src_y, mh,
                                       ln_postmix_g, ln_postmix_b,
                                       ln_preffn_g, ln_preffn_b,
                                       midx, midy, pffn);
  k_ffn1<<<dim3(8, 64), 256, 0, stream>>>(pffn, W_gate, W_up, hid);
  k_ffn2<<<64, 256, 0, stream>>>(hid, W_down, midx, midy,
                                 ln_postmoe_g, ln_postmoe_b, outx, outy);
}
